// Model_14199161881000
// MI455X (gfx1250) — matrix operand order repaired and checked by reading, not compiled, not run on hardware
//
#include <hip/hip_runtime.h>
#include <hip/hip_bf16.h>

typedef __attribute__((ext_vector_type(16))) _Float16 v16h;
typedef __attribute__((ext_vector_type(8)))  _Float16 v8h;
typedef __attribute__((ext_vector_type(8)))  float    v8f;
typedef int v4i_ __attribute__((vector_size(16)));   // matches builtin param type

#define N_NODES 20000
#define BGRAPH  128
#define EDGES   320000
#define FIN     768
#define RAWF    5000
#define RAWFP   5024      // 5000 padded to mult of 32
#define HID     256
#define CLS     4
#define CAT2    10000
#define CAT2P   10016     // 10000 padded to mult of 32

// ---------------- WMMA GEMM: C[M,N] = A[M,Kp](f16,row-major,lda) x Bt[N,Kp](f16,row-major,ldb)^T
// block = 256 threads (8 waves). Tile: BM=128 x BN=128; wave w owns rows w*16..w*16+15.
// Double-buffered LDS, one barrier per k-step, next-stage loads overlapped with WMMA.
#define BM 128
#define BN 128
#define BK 32
#define LDT 40                 // LDS stride in halves (conflict-free b128 reads)
#define ABUF (BM * LDT)        // 5120 halves per stage
#define BBUF (BN * LDT)

#if defined(__AMDGCN__) && __has_builtin(__builtin_amdgcn_global_load_async_to_lds_b128)
#define ASYNC_LDS 1
#endif

#ifdef ASYNC_LDS
static __device__ __forceinline__ void async_b128(const void* g, void* l)
{
    __builtin_amdgcn_global_load_async_to_lds_b128(
        (__attribute__((address_space(1))) v4i_*)(g),
        (__attribute__((address_space(3))) v4i_*)(l),
        0, 0);
}
static __device__ __forceinline__ void wait_async0()
{
#if __has_builtin(__builtin_amdgcn_s_wait_asynccnt)
    __builtin_amdgcn_s_wait_asynccnt(0);
#else
    asm volatile("s_wait_asynccnt 0x0" ::: "memory");
#endif
}
#endif

__global__ __launch_bounds__(256) void gemm_wmma_f16(
    const _Float16* __restrict__ A, int lda,
    const _Float16* __restrict__ Bt, int ldb,
    float* __restrict__ C, int M, int N, int Kp,
    const float* __restrict__ bias, const float* __restrict__ alpha, int act)
{
    __shared__ _Float16 As[2 * ABUF];
    __shared__ _Float16 Bs[2 * BBUF];
    const int tid  = threadIdx.x;
    const int wave = tid >> 5;
    const int lane = tid & 31;
    const int lm   = lane & 15;
    const int kbA  = (lane < 16) ? 0 : 8;    // A-frag interleave base (K 0-7/16-23 vs 8-15/24-31)
    const int kbB  = (lane < 16) ? 0 : 8;    // B-frag interleave base (K 0-7/16-23 vs 8-15/24-31)
    const int mbase = blockIdx.x * BM;
    const int nbase = blockIdx.y * BN;

    // cooperative-load geometry: 128 rows x 4 uint4-chunks = 512 chunks; 256 threads x 2
    const int rowc0 = tid >> 2;              // chunk r=0 -> rows 0..63
    const int rowc1 = (tid + 256) >> 2;      // chunk r=1 -> rows 64..127
    const int c4    = (tid & 3) * 8;         // half offset within the 32-half row segment

    // OOB rows: clamp to M-1 (their accumulators are discarded at the guarded store),
    // avoids exec-mask branches in the hot loop.
    int ra0 = mbase + rowc0; if (ra0 >= M) ra0 = M - 1;
    int ra1 = mbase + rowc1; if (ra1 >= M) ra1 = M - 1;
    const _Float16* gA0 = A + (size_t)ra0 * lda + c4;
    const _Float16* gA1 = A + (size_t)ra1 * lda + c4;
    const _Float16* gB0 = Bt + (size_t)(nbase + rowc0) * ldb + c4;
    const _Float16* gB1 = Bt + (size_t)(nbase + rowc1) * ldb + c4;
    _Float16* lA0 = &As[rowc0 * LDT + c4];
    _Float16* lA1 = &As[rowc1 * LDT + c4];
    _Float16* lB0 = &Bs[rowc0 * LDT + c4];
    _Float16* lB1 = &Bs[rowc1 * LDT + c4];

    v8f acc[8] = {};

#ifndef ASYNC_LDS
    uint4 ga0, ga1, gb0, gb1;
    auto loadTiles = [&](int k0) {
        ga0 = *(const uint4*)(gA0 + k0);
        ga1 = *(const uint4*)(gA1 + k0);
        gb0 = *(const uint4*)(gB0 + k0);
        gb1 = *(const uint4*)(gB1 + k0);
    };
    auto storeTiles = [&](int cur) {
        *(uint4*)(lA0 + cur * ABUF) = ga0;
        *(uint4*)(lA1 + cur * ABUF) = ga1;
        *(uint4*)(lB0 + cur * BBUF) = gb0;
        *(uint4*)(lB1 + cur * BBUF) = gb1;
    };
#else
    auto issueStage = [&](int k0, int cur) {
        async_b128(gA0 + k0, lA0 + cur * ABUF);
        async_b128(gA1 + k0, lA1 + cur * ABUF);
        async_b128(gB0 + k0, lB0 + cur * BBUF);
        async_b128(gB1 + k0, lB1 + cur * BBUF);
    };
#endif

    auto compute = [&](int cur) {
        const _Float16* Ab = &As[cur * ABUF];
        const _Float16* Bb = &Bs[cur * BBUF];
        int arow = wave * 16 + lm;
        v8h alo = *(const v8h*)(Ab + arow * LDT + kbA);
        v8h ahi = *(const v8h*)(Ab + arow * LDT + 16 + kbA);
        v16h a;
#pragma unroll
        for (int t = 0; t < 8; ++t) { a[t] = alo[t]; a[t + 8] = ahi[t]; }

        auto ldB = [&](int j) {
            int col = j * 16 + lm;
            v8h l = *(const v8h*)(Bb + col * LDT + kbB);
            v8h h = *(const v8h*)(Bb + col * LDT + kbB + 16);
            v16h r;
#pragma unroll
            for (int t = 0; t < 8; ++t) { r[t] = l[t]; r[t + 8] = h[t]; }
            return r;
        };
        // depth-2 fragment pipeline: fragment j+2 loads while WMMA j executes
        v16h f0 = ldB(0);
        v16h f1 = ldB(1);
#pragma unroll
        for (int j = 0; j < 8; ++j) {
            v16h f2 = f0;
            if (j < 6) f2 = ldB(j + 2);
            acc[j] = __builtin_amdgcn_wmma_f32_16x16x32_f16(
                false, a, false, f0, (short)0, acc[j], false, false);
            f0 = f1; f1 = f2;
        }
    };

#ifdef ASYNC_LDS
    issueStage(0, 0);
    int cur = 0;
    for (int k0 = 0; k0 < Kp; k0 += BK) {
        wait_async0();                           // own stage landed in LDS
        __syncthreads();                         // everyone's stage landed
        if (k0 + BK < Kp) issueStage(k0 + BK, cur ^ 1);  // DMA during WMMAs
        compute(cur);
        cur ^= 1;
    }
#else
    loadTiles(0);
    int cur = 0;
    for (int k0 = 0; k0 < Kp; k0 += BK) {
        storeTiles(cur);
        __syncthreads();
        if (k0 + BK < Kp) loadTiles(k0 + BK);    // in flight during WMMAs
        compute(cur);
        cur ^= 1;
    }
#endif

    // ---- epilogue: C layout: VGPR r, lanes 0-15 -> M=r, lanes 16-31 -> M=8+r; N=lm
    const int mhi = (lane < 16) ? 0 : 8;
    float al = (act == 1 && alpha) ? alpha[0] : 0.0f;
#pragma unroll
    for (int j = 0; j < 8; ++j) {
        int col = nbase + j * 16 + lm;
#pragma unroll
        for (int r = 0; r < 8; ++r) {
            int rowg = mbase + wave * 16 + mhi + r;
            if (rowg < M) {
                float v = acc[j][r];
                if (bias) v += bias[col];
                if (act == 1) v = (v >= 0.0f) ? v : al * v;
                C[(size_t)rowg * N + col] = v;
            }
        }
    }
}

// ---------------- elementwise / conversion kernels ----------------
__global__ void cvt_pad_f16(const float* __restrict__ src, _Float16* __restrict__ dst,
                            int M, int K, int Kp)
{
    int idx = blockIdx.x * blockDim.x + threadIdx.x;
    if (idx >= M * Kp) return;
    int row = idx / Kp, k = idx - row * Kp;
    dst[idx] = (k < K) ? (_Float16)src[(size_t)row * K + k] : (_Float16)0.0f;
}

// W[K,H] (f32) -> Wt[H,Kp] (f16), zero-padded K
__global__ void trans_cvt_f16(const float* __restrict__ W, _Float16* __restrict__ Wt,
                              int K, int H, int Kp)
{
    int idx = blockIdx.x * blockDim.x + threadIdx.x;
    if (idx >= H * Kp) return;
    int h = idx / Kp, k = idx - h * Kp;
    Wt[idx] = (k < K) ? (_Float16)W[(size_t)k * H + h] : (_Float16)0.0f;
}

__global__ void deg_init(float* deg, int n)
{
    int i = blockIdx.x * blockDim.x + threadIdx.x;
    if (i < n) deg[i] = 1.0f;   // self loop
}
__global__ void deg_accum(const int* __restrict__ dst, float* deg, int e)
{
    int i = blockIdx.x * blockDim.x + threadIdx.x;
    if (i < e) atomicAdd(&deg[dst[i]], 1.0f);
}
__global__ void make_dinv(float* deg, int n)
{
    int i = blockIdx.x * blockDim.x + threadIdx.x;
    if (i < n) deg[i] = rsqrtf(fmaxf(deg[i], 1.0f));
}

// out[i,f] = h[i,f] * dinv[i]^2   (self-loop term initializer)
__global__ void self_loop(const float* __restrict__ h, const float* __restrict__ dinv,
                          float* __restrict__ out, int total)
{
    int idx = blockIdx.x * blockDim.x + threadIdx.x;
    if (idx >= total) return;
    float w = dinv[idx >> 8];
    out[idx] = h[idx] * w * w;
}

// one block (256 threads = HID feats) per edge: out[d] += h[s]*dinv[s]*dinv[d]
__global__ void edge_scatter(const float* __restrict__ h, const int* __restrict__ src,
                             const int* __restrict__ dst, const float* __restrict__ dinv,
                             float* __restrict__ out)
{
    int e = blockIdx.x, f = threadIdx.x;
    int s = src[e], d = dst[e];
    float w = dinv[s] * dinv[d];
    atomicAdd(&out[(size_t)d * HID + f], h[(size_t)s * HID + f] * w);
}

__global__ void bias_relu(const float* __restrict__ x, const float* __restrict__ bias,
                          float* __restrict__ y, _Float16* __restrict__ y16, int total)
{
    int idx = blockIdx.x * blockDim.x + threadIdx.x;
    if (idx >= total) return;
    float v = fmaxf(x[idx] + bias[idx & (HID - 1)], 0.0f);
    y[idx] = v;
    if (y16) y16[idx] = (_Float16)v;
}

__global__ void pool_max(const float* __restrict__ y, const int* __restrict__ batch,
                         float* __restrict__ pool, int total)
{
    int idx = blockIdx.x * blockDim.x + threadIdx.x;
    if (idx >= total) return;
    int b = batch[idx >> 8];
    // values are post-ReLU (>=0): int compare == float compare
    atomicMax((int*)&pool[b * HID + (idx & (HID - 1))], __float_as_int(y[idx]));
}

__global__ void pool_sum(const float* __restrict__ y, const int* __restrict__ batch,
                         float* __restrict__ pool, int total)
{
    int idx = blockIdx.x * blockDim.x + threadIdx.x;
    if (idx >= total) return;
    int b = batch[idx >> 8];
    atomicAdd(&pool[b * HID + (idx & (HID - 1))], y[idx]);
}

__global__ void cnt_accum(const int* __restrict__ batch, float* cnt, int n)
{
    int i = blockIdx.x * blockDim.x + threadIdx.x;
    if (i < n) atomicAdd(&cnt[batch[i]], 1.0f);
}

__global__ void pool_div(float* pool, const float* __restrict__ cnt, int total)
{
    int idx = blockIdx.x * blockDim.x + threadIdx.x;
    if (idx >= total) return;
    pool[idx] /= fmaxf(cnt[idx >> 8], 1.0f);
}

__global__ void seg_bounds(const int* __restrict__ batch, int* s, int* e, int n)
{
    int i = blockIdx.x * blockDim.x + threadIdx.x;
    if (i >= n) return;
    int b = batch[i];
    atomicMin(&s[b], i);
    atomicMax(&e[b], i);
}

// segment mean of data (f16 copy, ld RAWFP) using contiguous per-graph row ranges
__global__ void mean_raw(const _Float16* __restrict__ d16, const int* __restrict__ s,
                         const int* __restrict__ e, float* __restrict__ newx)
{
    int b = blockIdx.x;
    int f = blockIdx.y * 256 + threadIdx.x;
    if (f >= RAWF) return;
    int i0 = s[b], i1 = e[b];
    float acc = 0.0f;
    int cnt = 0;
    for (int i = i0; i <= i1; ++i) { acc += (float)d16[(size_t)i * RAWFP + f]; ++cnt; }
    newx[(size_t)b * CAT2 + f] = (cnt > 0) ? acc / (float)cnt : 0.0f;
}

__global__ void root_gather(const float* __restrict__ data_x, const int* __restrict__ root,
                            float* __restrict__ newx)
{
    int b = blockIdx.x;
    int f = blockIdx.y * 256 + threadIdx.x;
    if (f >= RAWF) return;
    newx[(size_t)b * CAT2 + RAWF + f] = data_x[(size_t)root[b] * RAWF + f];
}

__global__ void concat5(const float* __restrict__ p0, const float* __restrict__ p1,
                        const float* __restrict__ p2, const float* __restrict__ nx,
                        const float* __restrict__ px, float* __restrict__ out5)
{
    int idx = blockIdx.x * blockDim.x + threadIdx.x;
    if (idx >= BGRAPH * 1280) return;
    int b = idx / 1280, j = idx - b * 1280;
    float v;
    if (j < 256)       v = p0[b * HID + j];
    else if (j < 512)  v = p1[b * HID + j - 256];
    else if (j < 768)  v = p2[b * HID + j - 512];
    else if (j < 1024) v = nx[b * HID + j - 768];
    else               v = px[b * HID + j - 1024];
    out5[idx] = v;
}

__global__ void final_logits(const float* __restrict__ out5, const float* __restrict__ W5,
                             const float* __restrict__ b5, float* __restrict__ out)
{
    int b = threadIdx.x;
    if (b >= BGRAPH) return;
    float z[CLS];
#pragma unroll
    for (int c = 0; c < CLS; ++c) z[c] = b5[c];
    for (int k = 0; k < 5 * HID; ++k) {
        float v = out5[(size_t)b * 5 * HID + k];
#pragma unroll
        for (int c = 0; c < CLS; ++c) z[c] += v * W5[k * CLS + c];
    }
    float m = z[0];
#pragma unroll
    for (int c = 1; c < CLS; ++c) m = fmaxf(m, z[c]);
    float ssum = 0.0f;
#pragma unroll
    for (int c = 0; c < CLS; ++c) ssum += expf(z[c] - m);
    float lse = m + logf(ssum);
#pragma unroll
    for (int c = 0; c < CLS; ++c) out[b * CLS + c] = z[c] - lse;
}

// ---------------- host side ----------------
static inline size_t alignup(size_t x) { return (x + 255) & ~(size_t)255; }

extern "C" void kernel_launch(void* const* d_in, const int* in_sizes, int n_in,
                              void* d_out, int out_size, void* d_ws, size_t ws_size,
                              hipStream_t stream)
{
    (void)in_sizes; (void)n_in; (void)out_size; (void)ws_size;
    const float* graph_x = (const float*)d_in[0];
    const float* data_x  = (const float*)d_in[2];
    const int*   ei      = (const int*)d_in[3];
    const int*   rei     = (const int*)d_in[4];
    const int*   gbatch  = (const int*)d_in[5];
    const int*   xbatch  = (const int*)d_in[6];
    const int*   root    = (const int*)d_in[7];
    const float* W1  = (const float*)d_in[8];   const float* b1  = (const float*)d_in[9];
    const float* Wc0 = (const float*)d_in[10];  const float* bc0 = (const float*)d_in[11];
    const float* Wc1 = (const float*)d_in[12];  const float* bc1 = (const float*)d_in[13];
    const float* Wc2 = (const float*)d_in[14];  const float* bc2 = (const float*)d_in[15];
    const float* Wl1 = (const float*)d_in[16];  const float* bl1 = (const float*)d_in[17];
    const float* Wl2 = (const float*)d_in[18];  const float* bl2 = (const float*)d_in[19];
    const float* alpha = (const float*)d_in[20];
    const float* W5  = (const float*)d_in[21];  const float* b5  = (const float*)d_in[22];

    const int* src1 = ei;        const int* dst1 = ei + EDGES;
    const int* srcr = rei;       const int* dstr = rei + EDGES;

    char* p = (char*)d_ws;
    auto take = [&](size_t bytes) -> void* { void* r = (void*)p; p += alignup(bytes); return r; };

    _Float16* A16d   = (_Float16*)take((size_t)N_NODES * RAWFP * 2);
    _Float16* A16g   = (_Float16*)take((size_t)N_NODES * FIN * 2);
    _Float16* Wt1    = (_Float16*)take((size_t)HID * FIN * 2);
    _Float16* Wtc0   = (_Float16*)take((size_t)HID * RAWFP * 2);
    _Float16* Wtc1   = (_Float16*)take((size_t)HID * HID * 2);
    _Float16* Wtc2   = (_Float16*)take((size_t)HID * HID * 2);
    _Float16* Wtl1   = (_Float16*)take((size_t)512 * CAT2P * 2);
    _Float16* Wtl2   = (_Float16*)take((size_t)HID * 512 * 2);
    float*    buf1   = (float*)take((size_t)N_NODES * HID * 4);
    float*    buf2   = (float*)take((size_t)N_NODES * HID * 4);
    _Float16* xone16 = (_Float16*)take((size_t)N_NODES * HID * 2);
    float*    dinv1  = (float*)take((size_t)N_NODES * 4);
    float*    dinvr  = (float*)take((size_t)N_NODES * 4);
    float*    poolx  = (float*)take((size_t)BGRAPH * HID * 4);
    float*    pool0  = (float*)take((size_t)BGRAPH * HID * 4);
    float*    pool1  = (float*)take((size_t)BGRAPH * HID * 4);
    float*    pool2  = (float*)take((size_t)BGRAPH * HID * 4);
    float*    cnt    = (float*)take((size_t)BGRAPH * 4);
    int*      segs   = (int*)take((size_t)BGRAPH * 4);
    int*      sege   = (int*)take((size_t)BGRAPH * 4);
    float*    newx   = (float*)take((size_t)BGRAPH * CAT2 * 4);
    _Float16* newx16 = (_Float16*)take((size_t)BGRAPH * CAT2P * 2);
    float*    lin1   = (float*)take((size_t)BGRAPH * 512 * 4);
    _Float16* lin116 = (_Float16*)take((size_t)BGRAPH * 512 * 2);
    float*    nxout  = (float*)take((size_t)BGRAPH * HID * 4);
    float*    out5   = (float*)take((size_t)BGRAPH * 5 * HID * 4);

    const int TPB = 256;
    auto g1 = [](long long n, int tpb) { return (unsigned)((n + tpb - 1) / tpb); };
    const int NH = N_NODES * HID;

    // per-call re-initialization (graph-replay safe: async memset nodes)
    (void)hipMemsetAsync(poolx, 0, (size_t)BGRAPH * HID * 4, stream);
    (void)hipMemsetAsync(pool0, 0, (size_t)BGRAPH * HID * 4, stream);
    (void)hipMemsetAsync(pool1, 0, (size_t)BGRAPH * HID * 4, stream);
    (void)hipMemsetAsync(pool2, 0, (size_t)BGRAPH * HID * 4, stream);
    (void)hipMemsetAsync(cnt,   0, (size_t)BGRAPH * 4, stream);
    (void)hipMemsetAsync(segs,  0x7f, (size_t)BGRAPH * 4, stream);   // large int
    (void)hipMemsetAsync(sege,  0xff, (size_t)BGRAPH * 4, stream);   // -1

    // ---- conversions / transposes
    cvt_pad_f16<<<g1((long long)N_NODES * RAWFP, TPB), TPB, 0, stream>>>(data_x, A16d, N_NODES, RAWF, RAWFP);
    cvt_pad_f16<<<g1((long long)N_NODES * FIN, TPB), TPB, 0, stream>>>(graph_x, A16g, N_NODES, FIN, FIN);
    trans_cvt_f16<<<g1((long long)HID * FIN, TPB), TPB, 0, stream>>>(W1, Wt1, FIN, HID, FIN);
    trans_cvt_f16<<<g1((long long)HID * RAWFP, TPB), TPB, 0, stream>>>(Wc0, Wtc0, RAWF, HID, RAWFP);
    trans_cvt_f16<<<g1((long long)HID * HID, TPB), TPB, 0, stream>>>(Wc1, Wtc1, HID, HID, HID);
    trans_cvt_f16<<<g1((long long)HID * HID, TPB), TPB, 0, stream>>>(Wc2, Wtc2, HID, HID, HID);
    trans_cvt_f16<<<g1((long long)512 * CAT2P, TPB), TPB, 0, stream>>>(Wl1, Wtl1, CAT2, 512, CAT2P);
    trans_cvt_f16<<<g1((long long)HID * 512, TPB), TPB, 0, stream>>>(Wl2, Wtl2, 512, HID, 512);

    // ---- degrees -> dinv (both edge sets)
    deg_init<<<g1(N_NODES, TPB), TPB, 0, stream>>>(dinv1, N_NODES);
    deg_accum<<<g1(EDGES, TPB), TPB, 0, stream>>>(dst1, dinv1, EDGES);
    make_dinv<<<g1(N_NODES, TPB), TPB, 0, stream>>>(dinv1, N_NODES);
    deg_init<<<g1(N_NODES, TPB), TPB, 0, stream>>>(dinvr, N_NODES);
    deg_accum<<<g1(EDGES, TPB), TPB, 0, stream>>>(dstr, dinvr, EDGES);
    make_dinv<<<g1(N_NODES, TPB), TPB, 0, stream>>>(dinvr, N_NODES);

    // ---- segment bounds + counts (batch is sorted)
    seg_bounds<<<g1(N_NODES, TPB), TPB, 0, stream>>>(xbatch, segs, sege, N_NODES);
    cnt_accum<<<g1(N_NODES, TPB), TPB, 0, stream>>>(xbatch, cnt, N_NODES);

    dim3 gemmBig((N_NODES + BM - 1) / BM, HID / BN);   // (157, 2)

    // ================= branch 1: GCN(graph_x) -> relu -> segment_max
    gemm_wmma_f16<<<gemmBig, 256, 0, stream>>>(A16g, FIN, Wt1, FIN, buf1,
                                               N_NODES, HID, FIN, nullptr, nullptr, 0);
    self_loop<<<g1(NH, TPB), TPB, 0, stream>>>(buf1, dinv1, buf2, NH);
    edge_scatter<<<EDGES, HID, 0, stream>>>(buf1, src1, dst1, dinv1, buf2);
    bias_relu<<<g1(NH, TPB), TPB, 0, stream>>>(buf2, b1, buf1, nullptr, NH);
    pool_max<<<g1(NH, TPB), TPB, 0, stream>>>(buf1, gbatch, poolx, NH);

    // ================= branch 3: 3x GCN(raw graph) with mean pools
    // layer 0 (the big GEMM: 20000x5024x256)
    gemm_wmma_f16<<<gemmBig, 256, 0, stream>>>(A16d, RAWFP, Wtc0, RAWFP, buf1,
                                               N_NODES, HID, RAWFP, nullptr, nullptr, 0);
    self_loop<<<g1(NH, TPB), TPB, 0, stream>>>(buf1, dinvr, buf2, NH);
    edge_scatter<<<EDGES, HID, 0, stream>>>(buf1, srcr, dstr, dinvr, buf2);
    bias_relu<<<g1(NH, TPB), TPB, 0, stream>>>(buf2, bc0, buf1, xone16, NH);
    pool_sum<<<g1(NH, TPB), TPB, 0, stream>>>(buf1, xbatch, pool0, NH);
    // layer 1
    gemm_wmma_f16<<<gemmBig, 256, 0, stream>>>(xone16, HID, Wtc1, HID, buf2,
                                               N_NODES, HID, HID, nullptr, nullptr, 0);
    self_loop<<<g1(NH, TPB), TPB, 0, stream>>>(buf2, dinvr, buf1, NH);
    edge_scatter<<<EDGES, HID, 0, stream>>>(buf2, srcr, dstr, dinvr, buf1);
    bias_relu<<<g1(NH, TPB), TPB, 0, stream>>>(buf1, bc1, buf2, xone16, NH);
    pool_sum<<<g1(NH, TPB), TPB, 0, stream>>>(buf2, xbatch, pool1, NH);
    // layer 2
    gemm_wmma_f16<<<gemmBig, 256, 0, stream>>>(xone16, HID, Wtc2, HID, buf1,
                                               N_NODES, HID, HID, nullptr, nullptr, 0);
    self_loop<<<g1(NH, TPB), TPB, 0, stream>>>(buf1, dinvr, buf2, NH);
    edge_scatter<<<EDGES, HID, 0, stream>>>(buf1, srcr, dstr, dinvr, buf2);
    bias_relu<<<g1(NH, TPB), TPB, 0, stream>>>(buf2, bc2, buf1, xone16, NH);
    pool_sum<<<g1(NH, TPB), TPB, 0, stream>>>(buf1, xbatch, pool2, NH);

    pool_div<<<g1(BGRAPH * HID, TPB), TPB, 0, stream>>>(pool0, cnt, BGRAPH * HID);
    pool_div<<<g1(BGRAPH * HID, TPB), TPB, 0, stream>>>(pool1, cnt, BGRAPH * HID);
    pool_div<<<g1(BGRAPH * HID, TPB), TPB, 0, stream>>>(pool2, cnt, BGRAPH * HID);

    // ================= branch 2: [mean || root] -> PReLU MLP
    dim3 mg(BGRAPH, (RAWF + 255) / 256);
    mean_raw<<<mg, 256, 0, stream>>>(A16d, segs, sege, newx);
    root_gather<<<mg, 256, 0, stream>>>(data_x, root, newx);
    cvt_pad_f16<<<g1((long long)BGRAPH * CAT2P, TPB), TPB, 0, stream>>>(newx, newx16, BGRAPH, CAT2, CAT2P);
    dim3 gl1((BGRAPH + BM - 1) / BM, 512 / BN);        // (1, 4)
    gemm_wmma_f16<<<gl1, 256, 0, stream>>>(newx16, CAT2P, Wtl1, CAT2P, lin1,
                                           BGRAPH, 512, CAT2P, bl1, alpha, 1);
    cvt_pad_f16<<<g1((long long)BGRAPH * 512, TPB), TPB, 0, stream>>>(lin1, lin116, BGRAPH, 512, 512);
    dim3 gl2((BGRAPH + BM - 1) / BM, HID / BN);        // (1, 2)
    gemm_wmma_f16<<<gl2, 256, 0, stream>>>(lin116, 512, Wtl2, 512, nxout,
                                           BGRAPH, HID, 512, bl2, alpha, 1);

    // ================= final: concat -> W5 -> log_softmax
    concat5<<<g1((long long)BGRAPH * 1280, TPB), TPB, 0, stream>>>(pool0, pool1, pool2, nxout, poolx, out5);
    final_logits<<<1, 128, 0, stream>>>(out5, W5, b5, (float*)d_out);
}
